// TransformerLearner_13082470383921
// MI455X (gfx1250) — matrix operand order repaired and checked by reading, not compiled, not run on hardware
//
#include <hip/hip_runtime.h>

// ---------------- problem constants (from reference) ----------------
constexpr int E_N   = 50000;
constexpr int R_N   = 22;
constexpr int NSTEP = 3;
constexpr int RANKN = 3;
constexpr int D     = 128;
constexpr int H     = 8;
constexpr int HD    = 16;
constexpr int FF    = 2048;
constexpr int NQ    = 24;
constexpr int LYR   = 6;
constexpr int B     = 32;
constexpr int NE    = 1000000;
constexpr int S     = 3;           // sequence length (NUM_STEP)
constexpr int M     = B * S;       // 96 GEMM rows
constexpr int RP1   = R_N + 1;     // 23
constexpr float LN_EPS = 1e-5f;
constexpr float THR    = 1e-20f;
constexpr size_t PRED_ELEMS = (size_t)B * E_N;  // 1,600,000

typedef __attribute__((ext_vector_type(16))) _Float16 v16h;
typedef __attribute__((ext_vector_type(8)))  float    v8f;

// ---------------------------------------------------------------
// WMMA 16x16 tile GEMM step:  C += A[tm-tile, :K] * W[tn-tile, :K]^T
// A: row-major f32 (LDS), lda stride.  W: row-major f32 [N][K] (global),
// i.e. B-matrix B[k][n] = W[n][k].  f32 -> f16 convert in the loaders,
// f32 accumulate in the matrix core.
// ---------------------------------------------------------------
__device__ __forceinline__ v8f wmma_tile_f16(const float* __restrict__ A, int lda,
                                             const float* __restrict__ W, int ldw,
                                             int tm, int tn, int K, v8f c) {
  const int lane = threadIdx.x & 31;
  const int mn   = lane & 15;   // A-frag row / B-frag col
  const int hs   = lane >> 4;
  for (int kk = 0; kk < K; kk += 32) {
    const float* abase = A + (size_t)(tm * 16 + mn) * lda + kk;
    const float* wbase = W + (size_t)(tn * 16 + mn) * ldw + kk;
    v16h a, b;
#pragma unroll
    for (int j = 0; j < 16; ++j) {           // ISA 16-bit A 16x32 layout
      int ka = ((j >> 3) << 4) + (hs << 3) + (j & 7);
      a[j] = (_Float16)abase[ka];
    }
#pragma unroll
    for (int j = 0; j < 16; ++j) {           // ISA 16-bit B 32x16 layout
      int kb = ((j >> 3) << 4) + (hs << 3) + (j & 7);
      b[j] = (_Float16)wbase[kb];
    }
    c = __builtin_amdgcn_wmma_f32_16x16x32_f16(false, a, false, b, (short)0, c,
                                               false, false);
  }
  return c;
}

__device__ __forceinline__ void layernorm_rows(float* x, const float* g,
                                               const float* bb, int tid) {
  if (tid < M) {                       // waves 0..2 fully active (uniform)
    float* row = x + tid * D;
    float m = 0.f;
    for (int d = 0; d < D; ++d) m += row[d];
    m *= (1.0f / D);
    float v = 0.f;
    for (int d = 0; d < D; ++d) { float t = row[d] - m; v += t * t; }
    v *= (1.0f / D);
    float inv = rsqrtf(v + LN_EPS);
    for (int d = 0; d < D; ++d) row[d] = (row[d] - m) * inv * g[d] + bb[d];
  }
}

// ---------------------------------------------------------------
// One workgroup per rank r: run all 6 encoder layers + softmax head.
// LDS: xbuf[96][128] (48KB) + sbuf[96][384] (144KB) = 192KB.
// ---------------------------------------------------------------
__global__ void __launch_bounds__(256)
k_transformer(const int* __restrict__ qq, const float* __restrict__ emb,
              const float* __restrict__ in_w, const float* __restrict__ in_b,
              const float* __restrict__ out_w, const float* __restrict__ out_b,
              const float* __restrict__ ln1_g, const float* __restrict__ ln1_b,
              const float* __restrict__ ln2_g, const float* __restrict__ ln2_b,
              const float* __restrict__ ff1_w, const float* __restrict__ ff1_b,
              const float* __restrict__ ff2_w, const float* __restrict__ ff2_b,
              const float* __restrict__ lin_w, const float* __restrict__ lin_b,
              float* __restrict__ atts_out) {
  extern __shared__ float lds[];
  float* xbuf = lds;            // [96][128]
  float* sbuf = lds + M * D;    // [96][384]  (qkv / attn-out / ff-hidden / logits)

  const int tid  = threadIdx.x;
  const int lane = tid & 31;
  const int wave = tid >> 5;    // 8 wave32's
  const int r    = blockIdx.x;

  // x0 = emb[queries]:  queries[b] = [qq[b], qq[b], NQ]
  for (int i = tid; i < M * D; i += blockDim.x) {
    int row = i / D, d = i % D;
    int b = row / S, s = row % S;
    int tok = (s < S - 1) ? qq[b] : NQ;
    xbuf[i] = emb[tok * D + d];
  }
  __syncthreads();

  for (int l = 0; l < LYR; ++l) {
    const size_t rl = (size_t)(r * LYR + l);
    const float* iw  = in_w  + rl * 3 * D * D;
    const float* ib  = in_b  + rl * 3 * D;
    const float* ow  = out_w + rl * D * D;
    const float* ob  = out_b + rl * D;
    const float* g1  = ln1_g + rl * D;   const float* b1 = ln1_b + rl * D;
    const float* g2  = ln2_g + rl * D;   const float* b2 = ln2_b + rl * D;
    const float* f1w = ff1_w + rl * FF * D;  const float* f1b = ff1_b + rl * FF;
    const float* f2w = ff2_w + rl * D * FF;  const float* f2b = ff2_b + rl * D;

    // ---- qkv = x @ iw^T + ib  -> sbuf[96][384]
    {
      const int TN = (3 * D) / 16;  // 24 col tiles, 6 row tiles -> 144 tiles
      for (int t = wave; t < 6 * TN; t += 8) {
        int tm = t / TN, tn = t % TN;
        v8f c = {};
        c = wmma_tile_f16(xbuf, D, iw, D, tm, tn, D, c);
        int n = lane & 15, hs = lane >> 4;
#pragma unroll
        for (int v = 0; v < 8; ++v) {
          int mm = tm * 16 + v + 8 * hs, nn = tn * 16 + n;
          sbuf[mm * (3 * D) + nn] = c[v] + ib[nn];
        }
      }
    }
    __syncthreads();

    // ---- attention: one thread per (b,h), S=3 so tiny; o overwrites q slice
    {
      int b = tid >> 3, h = tid & 7;
      const int qo = h * HD;
      float sc[S][S];
#pragma unroll
      for (int i = 0; i < S; ++i)
#pragma unroll
        for (int j = 0; j < S; ++j) {
          float acc = 0.f;
          for (int d = 0; d < HD; ++d)
            acc += sbuf[(b * S + i) * 384 + qo + d] *
                   sbuf[(b * S + j) * 384 + 128 + qo + d];
          sc[i][j] = acc * 0.25f;  // 1/sqrt(HD)
        }
      float o[S][HD];
#pragma unroll
      for (int i = 0; i < S; ++i) {
        float mx = fmaxf(sc[i][0], fmaxf(sc[i][1], sc[i][2]));
        float e0 = __expf(sc[i][0] - mx);
        float e1 = __expf(sc[i][1] - mx);
        float e2 = __expf(sc[i][2] - mx);
        float inv = 1.f / (e0 + e1 + e2);
        float a0 = e0 * inv, a1 = e1 * inv, a2 = e2 * inv;
        for (int d = 0; d < HD; ++d)
          o[i][d] = a0 * sbuf[(b * S + 0) * 384 + 256 + qo + d] +
                    a1 * sbuf[(b * S + 1) * 384 + 256 + qo + d] +
                    a2 * sbuf[(b * S + 2) * 384 + 256 + qo + d];
      }
#pragma unroll
      for (int i = 0; i < S; ++i)
        for (int d = 0; d < HD; ++d)
          sbuf[(b * S + i) * 384 + qo + d] = o[i][d];  // own slice only
    }
    __syncthreads();

    // ---- x = x + o @ ow^T + ob   (A = sbuf with stride 384, first 128 cols)
    {
      const int TN = D / 16;  // 48 tiles
      for (int t = wave; t < 6 * TN; t += 8) {
        int tm = t / TN, tn = t % TN;
        v8f c = {};
        c = wmma_tile_f16(sbuf, 384, ow, D, tm, tn, D, c);
        int n = lane & 15, hs = lane >> 4;
#pragma unroll
        for (int v = 0; v < 8; ++v) {
          int mm = tm * 16 + v + 8 * hs, nn = tn * 16 + n;
          xbuf[mm * D + nn] += c[v] + ob[nn];
        }
      }
    }
    __syncthreads();
    layernorm_rows(xbuf, g1, b1, tid);
    __syncthreads();

    // ---- FF fused over 16 chunks of 128; out-accumulators live in VGPRs
    {
      const int TN = D / 16;  // 48 output tiles -> 6 per wave, fixed mapping
      v8f acc[6];
#pragma unroll
      for (int i = 0; i < 6; ++i) { v8f z = {}; acc[i] = z; }
      for (int ch = 0; ch < FF / D; ++ch) {
        // h_chunk = relu(x @ f1w[ch]^T + f1b[ch]) -> sbuf[96][128]
        for (int t = wave; t < 48; t += 8) {
          int tm = t / TN, tn = t % TN;
          v8f c = {};
          c = wmma_tile_f16(xbuf, D, f1w + (size_t)ch * D * D, D, tm, tn, D, c);
          int n = lane & 15, hs = lane >> 4;
#pragma unroll
          for (int v = 0; v < 8; ++v) {
            int mm = tm * 16 + v + 8 * hs, nn = tn * 16 + n;
            sbuf[mm * D + nn] = fmaxf(c[v] + f1b[ch * D + nn], 0.f);
          }
        }
        __syncthreads();
        // acc += h_chunk @ f2w[:, ch*128 : ch*128+128]^T  (B[k][n]=f2w[n][ch*128+k])
        {
          int i = 0;
          for (int t = wave; t < 48; t += 8, ++i) {
            int tm = t / TN, tn = t % TN;
            acc[i] = wmma_tile_f16(sbuf, D, f2w + ch * D, FF, tm, tn, D, acc[i]);
          }
        }
        __syncthreads();
      }
      // x = x + ff_out + f2b
      {
        int i = 0;
        for (int t = wave; t < 48; t += 8, ++i) {
          int tm = t / TN, tn = t % TN;
          int n = lane & 15, hs = lane >> 4;
#pragma unroll
          for (int v = 0; v < 8; ++v) {
            int mm = tm * 16 + v + 8 * hs, nn = tn * 16 + n;
            xbuf[mm * D + nn] += acc[i][v] + f2b[nn];
          }
        }
      }
    }
    __syncthreads();
    layernorm_rows(xbuf, g2, b2, tid);
    __syncthreads();
  }

  // ---- head: atts[r] = softmax(x @ lin_w^T + lin_b) over 23 outputs
  for (int idx = tid; idx < M * RP1; idx += blockDim.x) {
    int row = idx / RP1, j = idx % RP1;
    float acc = lin_b[j];
    for (int d = 0; d < D; ++d) acc += xbuf[row * D + d] * lin_w[j * D + d];
    sbuf[idx] = acc;
  }
  __syncthreads();
  if (tid < M) {
    int row = tid, b = row / S, s = row % S;
    float mx = -1e30f;
    for (int j = 0; j < RP1; ++j) mx = fmaxf(mx, sbuf[row * RP1 + j]);
    float ex[RP1];
    float sum = 0.f;
    for (int j = 0; j < RP1; ++j) { ex[j] = __expf(sbuf[row * RP1 + j] - mx); sum += ex[j]; }
    float inv = 1.f / sum;
    for (int j = 0; j < RP1; ++j)
      atts_out[((size_t)(r * B + b) * S + s) * RP1 + j] = ex[j] * inv;
  }
}

// ---------------- sparse propagation kernels ----------------
// memory / combined layout: [E][B] (B=32 contiguous floats per entity row)

__global__ void k_zero(float* __restrict__ p, int n) {
  for (int i = blockIdx.x * blockDim.x + threadIdx.x; i < n;
       i += gridDim.x * blockDim.x)
    p[i] = 0.f;
}

__global__ void k_seed(const int* __restrict__ tt, float* __restrict__ mem) {
  int b = threadIdx.x;
  if (b < B) mem[(size_t)tt[b] * B + b] = 1.f;  // one_hot tails
}

// combined[e][b] = memory[e][b] * att[b][R]   (self-transition term)
__global__ void k_selfterm(const float* __restrict__ mem,
                           const float* __restrict__ attp,  // + b*S*RP1 + j
                           float* __restrict__ comb) {
  for (int i = blockIdx.x * blockDim.x + threadIdx.x; i < E_N * B;
       i += gridDim.x * blockDim.x) {
    int b = i & 31;
    comb[i] = mem[i] * attp[b * (S * RP1) + R_N];
  }
}

// Fused gather/segment_sum/einsum:
//   combined[b][src] += val * att[b][rel] * memory[b][dst]
// One edge per wave32; lane = batch index b. 128B coalesced gather + FP32
// L2 atomic scatter. att table staged in LDS (stride 23 -> conflict-free).
__global__ void __launch_bounds__(256)
k_edges(const int* __restrict__ src, const int* __restrict__ dst,
        const int* __restrict__ rel, const float* __restrict__ val,
        const float* __restrict__ attp, const float* __restrict__ mem,
        float* __restrict__ comb) {
  __shared__ float s_att[B * RP1];
  for (int i = threadIdx.x; i < B * RP1; i += blockDim.x) {
    int b = i / RP1, j = i % RP1;
    s_att[i] = attp[b * (S * RP1) + j];
  }
  __syncthreads();
  const int lane = threadIdx.x & 31;
  const int wid  = (blockIdx.x * blockDim.x + threadIdx.x) >> 5;
  const int nw   = (gridDim.x * blockDim.x) >> 5;
  for (int e = wid; e < NE; e += nw) {
    int e2 = e + nw;
    if (e2 < NE) {  // prefetch next edge record into cache
      __builtin_prefetch(&src[e2], 0, 1);
      __builtin_prefetch(&dst[e2], 0, 1);
      __builtin_prefetch(&rel[e2], 0, 1);
      __builtin_prefetch(&val[e2], 0, 1);
    }
    int s = src[e], d = dst[e], rl = rel[e];
    float v = val[e];
    float m = mem[(size_t)d * B + lane];
    float contrib = v * s_att[lane * RP1 + rl] * m;
    __hip_atomic_fetch_add(&comb[(size_t)s * B + lane], contrib,
                           __ATOMIC_RELAXED, __HIP_MEMORY_SCOPE_AGENT);
  }
}

// denom[b] = sum_e |combined[e][b]|  (block partials + 32 atomics)
__global__ void __launch_bounds__(256)
k_denom(const float* __restrict__ comb, float* __restrict__ denom) {
  __shared__ float red[256];
  const int tid = threadIdx.x;
  float acc = 0.f;
  for (int i = blockIdx.x * blockDim.x + tid; i < E_N * B;
       i += gridDim.x * blockDim.x)
    acc += fabsf(comb[i]);  // i & 31 == tid & 31 (stride multiple of 32)
  red[tid] = acc;
  __syncthreads();
  if (tid < 32) {
    float s = 0.f;
    for (int k = tid; k < 256; k += 32) s += red[k];
    __hip_atomic_fetch_add(&denom[tid], s, __ATOMIC_RELAXED,
                           __HIP_MEMORY_SCOPE_AGENT);
  }
}

// memory = combined / max(denom, THR); on the last step also pred += memory
__global__ void k_norm(const float* __restrict__ comb,
                       const float* __restrict__ denom,
                       float* __restrict__ mem, float* __restrict__ pred,
                       int addpred) {
  for (int i = blockIdx.x * blockDim.x + threadIdx.x; i < E_N * B;
       i += gridDim.x * blockDim.x) {
    int b = i & 31, e = i >> 5;
    float m = comb[i] / fmaxf(denom[b], THR);
    mem[i] = m;
    if (addpred) pred[(size_t)b * E_N + e] += m;
  }
}

// ---------------------------------------------------------------
extern "C" void kernel_launch(void* const* d_in, const int* in_sizes, int n_in,
                              void* d_out, int out_size, void* d_ws,
                              size_t ws_size, hipStream_t stream) {
  const int*   qq    = (const int*)d_in[0];
  const int*   tt    = (const int*)d_in[1];
  const int*   esrc  = (const int*)d_in[2];
  const int*   edst  = (const int*)d_in[3];
  const int*   erel  = (const int*)d_in[4];
  const float* evalv = (const float*)d_in[5];
  const float* emb   = (const float*)d_in[6];
  const float* in_w  = (const float*)d_in[7];
  const float* in_b  = (const float*)d_in[8];
  const float* out_w = (const float*)d_in[9];
  const float* out_b = (const float*)d_in[10];
  const float* ln1_g = (const float*)d_in[11];
  const float* ln1_b = (const float*)d_in[12];
  const float* ln2_g = (const float*)d_in[13];
  const float* ln2_b = (const float*)d_in[14];
  const float* ff1_w = (const float*)d_in[15];
  const float* ff1_b = (const float*)d_in[16];
  const float* ff2_w = (const float*)d_in[17];
  const float* ff2_b = (const float*)d_in[18];
  const float* lin_w = (const float*)d_in[19];
  const float* lin_b = (const float*)d_in[20];

  float* pred = (float*)d_out;              // [B][E]
  float* atts = pred + PRED_ELEMS;          // [RANK][B][S][R+1]

  // workspace: memory [E][B], combined [E][B], denom [B]  (~12.2 MB)
  float* mem   = (float*)d_ws;
  float* comb  = mem + (size_t)E_N * B;
  float* denom = comb + (size_t)E_N * B;

  // ---- phase 1: transformer stacks -> atts (one WGP per rank)
  const size_t ldsBytes = (size_t)(M * D + M * 3 * D) * sizeof(float);  // 192KB
  k_transformer<<<RANKN, 256, ldsBytes, stream>>>(
      qq, emb, in_w, in_b, out_w, out_b, ln1_g, ln1_b, ln2_g, ln2_b, ff1_w,
      ff1_b, ff2_w, ff2_b, lin_w, lin_b, atts);

  // ---- phase 2: sparse propagation
  k_zero<<<1024, 256, 0, stream>>>(pred, (int)PRED_ELEMS);
  for (int r = 0; r < RANKN; ++r) {
    k_zero<<<1024, 256, 0, stream>>>(mem, E_N * B);
    k_seed<<<1, 64, 0, stream>>>(tt, mem);
    for (int t = 0; t < NSTEP - 1; ++t) {
      const float* attp = atts + (size_t)(r * B * S + t) * RP1;
      k_selfterm<<<1024, 256, 0, stream>>>(mem, attp, comb);
      k_zero<<<1, 64, 0, stream>>>(denom, B);
      k_edges<<<2048, 256, 0, stream>>>(esrc, edst, erel, evalv, attp, mem, comb);
      k_denom<<<1024, 256, 0, stream>>>(comb, denom);
      k_norm<<<1024, 256, 0, stream>>>(comb, denom, mem, pred,
                                       (t == NSTEP - 2) ? 1 : 0);
    }
  }
}
